// KANLayer_5239860101393
// MI455X (gfx1250) — matrix operand order repaired and checked by reading, not compiled, not run on hardware
//
#include <hip/hip_runtime.h>
#include <hip/hip_bf16.h>
#include <stdint.h>

typedef __attribute__((ext_vector_type(16))) _Float16 v16h;
typedef __attribute__((ext_vector_type(8)))  float    v8f;

#define B_DIM   4096
#define IN_DIM  512
#define OUT_DIM 512
#define NSEG    9
#define NPOW    4
#define KSEG    (IN_DIM * NPOW)        // 2048, K-span of one segment
#define KTOT    (NSEG * KSEG)          // 18432 total GEMM K
#define BM      64
#define BN      128
#define BK      64                     // two 32-wide WMMA windows per barrier pair
#define LDT     72                     // LDS row stride in halves: 144 B, 16B-aligned, 36 banks

static __device__ __forceinline__ uint64_t pack4h(float a, float b, float c, float d) {
  union { _Float16 h; unsigned short u; } ua, ub, uc, ud;
  ua.h = (_Float16)a; ub.h = (_Float16)b; uc.h = (_Float16)c; ud.h = (_Float16)d;
  return (uint64_t)ua.u | ((uint64_t)ub.u << 16) | ((uint64_t)uc.u << 32) | ((uint64_t)ud.u << 48);
}

// ---- kernel 1: coeff f32 [OUT][IN][S][P] -> f16 Bmat [OUT][k], k = s*KSEG + i*4 + p
__global__ __launch_bounds__(256) void kan_prep_coeff(const float* __restrict__ coeff,
                                                      uint64_t* __restrict__ Bh) {
  int idx = blockIdx.x * 256 + threadIdx.x;        // over OUT*IN*NSEG cells of 4 floats
  if (idx >= OUT_DIM * IN_DIM * NSEG) return;
  int j = idx / (IN_DIM * NSEG);
  int r = idx % (IN_DIM * NSEG);
  int i = r / NSEG;
  int s = r % NSEG;
  float4 c = *(reinterpret_cast<const float4*>(coeff) + idx);   // contiguous [j][i][s][0..3]
  Bh[(size_t)j * (KTOT / 4) + (size_t)s * (KSEG / 4) + i] = pack4h(c.x, c.y, c.z, c.w);
}

// ---- kernel 2: per (b,i): seg byte + packed f16 powers {1, t, t^2, t^3}
__global__ __launch_bounds__(256) void kan_prep_x(const float* __restrict__ x,
                                                  uint64_t* __restrict__ tp4,
                                                  unsigned char* __restrict__ segb) {
  int idx = blockIdx.x * 256 + threadIdx.x;        // over B*IN
  if (idx >= B_DIM * IN_DIM) return;
  float xc = fminf(fmaxf(x[idx], -1.f), 1.f);
  float u  = (xc + 1.f) * 4.5f;                    // h = 2/9 uniform knots
  int   s  = (int)floorf(u);
  s = s < 0 ? 0 : (s > NSEG - 1 ? NSEG - 1 : s);
  float t  = u - (float)s;
  float t2 = t * t;
  tp4[idx]  = pack4h(1.f, t, t2, t2 * t);
  segb[idx] = (unsigned char)s;
}

// ---- kernel 3: WMMA GEMM; masked expanded-A built in LDS per 64-wide K block
__global__ __launch_bounds__(256) void kan_gemm(const uint64_t* __restrict__ tp4,
                                                const unsigned char* __restrict__ segb,
                                                const uint64_t* __restrict__ Bh,
                                                const float* __restrict__ scale,
                                                float* __restrict__ y) {
  __shared__ _Float16 Atile[BM * LDT];             //  9,216 B
  __shared__ _Float16 Btile[BN * LDT];             // 18,432 B

  const int tid  = threadIdx.x;
  const int lane = tid & 31;
  const int wave = tid >> 5;                       // 0..7, arranged 2 (rows) x 4 (cols)
  const int b0   = blockIdx.x * BM;
  const int j0   = blockIdx.y * BN;

  const int mrow0 = (wave & 1) * 2;                // subtile rows {mrow0, mrow0+1} of 4
  const int jc0   = (wave >> 1) * 2;               // subtile cols {jc0, jc0+1} of 8

  v8f acc[2][2] = {};

  // staging roles
  const int arow = tid >> 2;                       // 0..63   A row; 4 inputs per thread
  const int aig  = tid & 3;                        //          i-quad within 16-input block
  const int brow = tid >> 1;                       // 0..127  B row; 64 B per thread
  const int bhlf = tid & 1;

  // fragment lane addressing in contiguous k order (k = 16*half + s), same for A and B (wave32, 16-bit operands)
  const int half16 = lane >> 4;
  const int l16    = lane & 15;
  const int kaA = half16 ? 16 : 0;                 // A: lanes<16 -> K 0-7 & 8-15
  const int kbA = half16 ? 24 : 8;                 //    lanes>=16 -> K 16-23 & 24-31
  const int kaB = half16 ? 16 : 0;                 // B: lane half splits K in half

  for (int k0 = 0; k0 < KTOT; k0 += BK) {
    const int s    = k0 >> 11;                     // 64-wide block lies in one segment
    const int iwin = (k0 & (KSEG - 1)) >> 2;       // first of 16 inputs in this block

    // --- build masked A tile: A[row][(i-iwin)*4 + p] = (seg==s) ? t^p : 0
    {
      const size_t base = (size_t)(b0 + arow) * IN_DIM + iwin + aig * 4;
      uint32_t sgw = *(const uint32_t*)(segb + base);     // 4 seg bytes
      #pragma unroll
      for (int q = 0; q < 4; ++q) {
        uint64_t tp = tp4[base + q];
        int sg = (sgw >> (8 * q)) & 0xff;
        *(uint64_t*)&Atile[arow * LDT + (aig * 4 + q) * 4] = (sg == s) ? tp : 0ull;
      }
    }
    // --- stage B tile (64 B per thread) from L2-resident f16 coeff
    {
      const uint4* src = (const uint4*)(Bh + (size_t)(j0 + brow) * (KTOT / 4)
                                           + (k0 >> 2) + bhlf * 8);
      uint4* dst = (uint4*)&Btile[brow * LDT + bhlf * 32];
      uint4 d0 = src[0], d1 = src[1], d2 = src[2], d3 = src[3];
      dst[0] = d0; dst[1] = d1; dst[2] = d2; dst[3] = d3;
      if (k0 + BK < KTOT)
        __builtin_prefetch(src + 4, 0, 1);         // next block -> global_prefetch_b8
    }
    __syncthreads();

    #pragma unroll
    for (int w = 0; w < 2; ++w) {                  // two 32-wide WMMA windows
      const int kb = w * 32;
      v16h afrag[2], bfrag[2];
      #pragma unroll
      for (int r = 0; r < 2; ++r) {
        const _Float16* base = &Atile[((mrow0 + r) * 16 + l16) * LDT + kb];
        *(uint4*)&afrag[r]        = *(const uint4*)(base + kaA);
        *(((uint4*)&afrag[r]) + 1) = *(const uint4*)(base + kbA);
      }
      #pragma unroll
      for (int c = 0; c < 2; ++c) {
        const _Float16* base = &Btile[((jc0 + c) * 16 + l16) * LDT + kb + kaB];
        *(uint4*)&bfrag[c]        = *(const uint4*)(base);
        *(((uint4*)&bfrag[c]) + 1) = *(const uint4*)(base + 8);
      }
      #pragma unroll
      for (int r = 0; r < 2; ++r)
        #pragma unroll
        for (int c = 0; c < 2; ++c)
          acc[r][c] = __builtin_amdgcn_wmma_f32_16x16x32_f16(
              false, afrag[r], false, bfrag[c], (short)0, acc[r][c], false, false);
    }
    __syncthreads();
  }

  // --- epilogue: C layout = 8 VGPRs; lanes 0-15: M=v, lanes 16-31: M=8+v
  const int rbase = half16 * 8;
  #pragma unroll
  for (int c = 0; c < 2; ++c) {
    int col = j0 + (jc0 + c) * 16 + l16;
    float sc = scale[col];
    #pragma unroll
    for (int r = 0; r < 2; ++r) {
      int row0 = b0 + (mrow0 + r) * 16 + rbase;
      #pragma unroll
      for (int v = 0; v < 8; ++v)
        y[(size_t)(row0 + v) * OUT_DIM + col] = acc[r][c][v] * sc;
    }
  }
}

extern "C" void kernel_launch(void* const* d_in, const int* in_sizes, int n_in,
                              void* d_out, int out_size, void* d_ws, size_t ws_size,
                              hipStream_t stream) {
  const float* x     = (const float*)d_in[0];   // [B, IN]
  const float* coeff = (const float*)d_in[1];   // [OUT, IN, S, P]
  const float* scale = (const float*)d_in[2];   // [OUT]
  float*       y     = (float*)d_out;           // [B, OUT]

  // workspace layout (bytes): Bh f16 coeff | tp4 packed powers | seg bytes
  const size_t bh_bytes  = (size_t)OUT_DIM * KTOT * sizeof(_Float16);   // 18,874,368
  const size_t tp4_bytes = (size_t)B_DIM * IN_DIM * sizeof(uint64_t);   // 16,777,216
  uint64_t*      Bh   = (uint64_t*)d_ws;
  uint64_t*      tp4  = (uint64_t*)((char*)d_ws + bh_bytes);
  unsigned char* segb = (unsigned char*)((char*)d_ws + bh_bytes + tp4_bytes);

  {
    int n = OUT_DIM * IN_DIM * NSEG;             // cells of 4 coefficients
    kan_prep_coeff<<<(n + 255) / 256, 256, 0, stream>>>(coeff, Bh);
  }
  {
    int n = B_DIM * IN_DIM;
    kan_prep_x<<<(n + 255) / 256, 256, 0, stream>>>(x, tp4, segb);
  }
  {
    dim3 grid(B_DIM / BM, OUT_DIM / BN);         // (64, 4)
    kan_gemm<<<grid, 256, 0, stream>>>(tp4, segb, Bh, scale, y);
  }
}
